// RetNet_42657615184544
// MI455X (gfx1250) — matrix operand order repaired and checked by reading, not compiled, not run on hardware
//
#include <hip/hip_runtime.h>
#include <hip/hip_bf16.h>
#include <math.h>
#include <stdint.h>

// ---------------- problem dims (fixed by reference) ----------------
#define L_LAYERS 2
#define NHEADS   8
#define DMODEL   1024
#define FFNDIM   4096
#define HEADDIM  128
#define BATCH    2
#define SEQ      2048
#define MROWS    (BATCH * SEQ)     // 4096

// ---------------- CDNA5 WMMA types ----------------
typedef __attribute__((ext_vector_type(16))) __bf16 v16bf;
typedef __attribute__((ext_vector_type(8)))  float  v8f;

// Native bf16 converts (lowers to v_fma_mixlo_bf16 / v_cvt path)
__device__ __forceinline__ unsigned int pack_bf16(float lo, float hi) {
  union { unsigned int u; __bf16 h[2]; } p;
  p.h[0] = (__bf16)lo;
  p.h[1] = (__bf16)hi;
  return p.u;
}
__device__ __forceinline__ unsigned short bf16bits(float f) {
  union { unsigned short u; __bf16 h; } p;
  p.h = (__bf16)f;
  return p.u;
}

// Generic pointer to LDS: low 32 bits are the LDS byte address.
__device__ __forceinline__ unsigned lds_off_of(const void* p) {
  return (unsigned)(unsigned long long)p;
}

// Async global->LDS 16B copy (ASYNCcnt-tracked, bypasses VGPRs).
__device__ __forceinline__ void async_ld128(unsigned lds_byte, const void* g) {
  asm volatile("global_load_async_to_lds_b128 %0, %1, off"
               :: "v"(lds_byte), "v"(g)
               : "memory");
}
__device__ __forceinline__ void async_wait0() {
  asm volatile("s_wait_asynccnt 0" ::: "memory");
}

// A-matrix 16x32 bf16 fragment (contiguous k order, not ISA table order): rows are packed K-pair dwords.
// lanes 0-15: M=lane, dwords 0..7; lanes 16-31: dwords 8..15.
__device__ __forceinline__ v16bf frag_A(const unsigned int* tile, int rowWords,
                                        int kw, int lane) {
  const unsigned int* row = tile + (lane & 15) * rowWords + kw;
  const int kb2 = (lane >> 4) << 3;
  union { unsigned int u[8]; v16bf v; } f;
#pragma unroll
  for (int i = 0; i < 4; ++i) f.u[i] = row[kb2 + i];
#pragma unroll
  for (int i = 4; i < 8; ++i) f.u[i] = row[kb2 + i];
  return f.v;
}

// B-matrix 32x16 bf16 fragment from a K-transposed tile [n][k-pair].
// lanes 0-15: N=lane, K=0..15 (dwords 0..7); lanes 16-31: K=16..31 (8..15).
__device__ __forceinline__ v16bf frag_B(const unsigned int* tileT, int rowWords,
                                        int kw, int lane) {
  const unsigned int* row = tileT + (lane & 15) * rowWords + kw;
  const int pb = (lane >> 4) << 3;
  union { unsigned int u[8]; v16bf v; } f;
#pragma unroll
  for (int i = 0; i < 8; ++i) f.u[i] = row[pb + i];
  return f.v;
}

__device__ __forceinline__ v8f wmma_bf16(v16bf a, v16bf b, v8f c) {
  return __builtin_amdgcn_wmma_f32_16x16x32_bf16(false, a, false, b,
                                                 (short)0, c, false, false);
}

__device__ __forceinline__ float silu(float x) {
  return x / (1.f + __expf(-x));
}

// =====================================================================
// bf16-WMMA GEMM on pre-packed operands.
// Apk: [M][K/2] dwords (bf16 pairs along K), ldaw = dwords per row.
// Bpk: [N][K/2] dwords (pre-transposed weights), ldbw = dwords per row.
// C[M,N] (f32) = act(A*B + bias) + Resid. Block 128x128, 8 waves,
// BK=32, double-buffered LDS staged by async-to-LDS DMA.
// Batch: A += (bz/adiv)*strideA, B += (bz%bmod)*strideB, C += bz*strideC.
// =====================================================================
template <int BIAS, int ACT_SILU, int RES>
__global__ __launch_bounds__(256) void gemm_pk(
    const unsigned int* __restrict__ Apk, int64_t strideA, int adiv, int ldaw,
    const unsigned int* __restrict__ Bpk, int64_t strideB, int bmod, int ldbw,
    const float* __restrict__ bias,
    const float* __restrict__ Resid, int64_t strideR,
    float* __restrict__ C, int64_t strideC,
    int M, int N, int K, int ldc) {
  __shared__ unsigned int lsA[2][128 * 16];   // [row][K-pair]  2 x 8 KB
  __shared__ unsigned int lsBT[2][128 * 16];  // [col][K-pair]  2 x 8 KB

  const int bz = blockIdx.z;
  Apk += (int64_t)(bz / adiv) * strideA;
  Bpk += (int64_t)(bz % bmod) * strideB;
  C += (int64_t)bz * strideC;
  if (RES) Resid += (int64_t)bz * strideR;

  const int tid = threadIdx.x;
  const int lane = tid & 31, wave = tid >> 5;
  const int m0 = blockIdx.y * 128, n0 = blockIdx.x * 128;
  const int wm = (wave & 3) * 32;   // two 16-row A fragments
  const int wn = (wave >> 2) * 64;  // four 16-col B tiles

  const unsigned aoff[2] = {lds_off_of(&lsA[0][0]), lds_off_of(&lsA[1][0])};
  const unsigned boff[2] = {lds_off_of(&lsBT[0][0]), lds_off_of(&lsBT[1][0])};

  // stage a 128-row x 16-dword tile: 512 x 16B async transfers, 2/thread
  auto stage = [&](const unsigned int* rowbase, int64_t ldw, unsigned ldsbyte) {
#pragma unroll
    for (int pass = 0; pass < 2; ++pass) {
      int idx = tid + pass * 256;
      int r = idx >> 2, ch = (idx & 3) << 2;  // dword offset within row
      async_ld128(ldsbyte + (unsigned)((r * 16 + ch) << 2),
                  rowbase + (int64_t)r * ldw + ch);
    }
  };

  v8f acc0[4] = {};  // rows wm..wm+15
  v8f acc1[4] = {};  // rows wm+16..wm+31

  stage(Apk + (int64_t)m0 * ldaw, ldaw, aoff[0]);
  stage(Bpk + (int64_t)n0 * ldbw, ldbw, boff[0]);
  async_wait0();
  __syncthreads();

  int buf = 0;
  for (int k0 = 0; k0 < K; k0 += 32) {
    const int nk = k0 + 32;
    if (nk < K) {  // async-DMA next K-tile into the other LDS buffer
      stage(Apk + (int64_t)m0 * ldaw + (nk >> 1), ldaw, aoff[buf ^ 1]);
      stage(Bpk + (int64_t)n0 * ldbw + (nk >> 1), ldbw, boff[buf ^ 1]);
    }
    if (k0 + 64 < K) {  // L2 prefetch two K-tiles ahead
      __builtin_prefetch(Apk + (int64_t)(m0 + (tid >> 1)) * ldaw +
                             ((k0 + 64) >> 1) + ((tid & 1) << 3),
                         0, 1);
      __builtin_prefetch(Bpk + (int64_t)(n0 + (tid >> 1)) * ldbw +
                             ((k0 + 64) >> 1) + ((tid & 1) << 3),
                         0, 1);
    }

    v16bf a0 = frag_A(lsA[buf] + wm * 16, 16, 0, lane);
    v16bf a1 = frag_A(lsA[buf] + (wm + 16) * 16, 16, 0, lane);
#pragma unroll
    for (int j = 0; j < 4; ++j) {
      v16bf bj = frag_B(lsBT[buf] + (wn + j * 16) * 16, 16, 0, lane);
      acc0[j] = wmma_bf16(a0, bj, acc0[j]);
      acc1[j] = wmma_bf16(a1, bj, acc1[j]);
    }
    async_wait0();
    __syncthreads();
    buf ^= 1;
  }

  // epilogue: C/D layout: VGPR v -> row v + 8*(lane>=16), col = lane&15
  const int lanecol = lane & 15;
  const int rsub = (lane >> 4) << 3;
#pragma unroll
  for (int half = 0; half < 2; ++half) {
    const int rbase = m0 + wm + half * 16 + rsub;
#pragma unroll
    for (int j = 0; j < 4; ++j) {
      const int col = n0 + wn + j * 16 + lanecol;
#pragma unroll
      for (int v = 0; v < 8; ++v) {
        float x = half ? acc1[j][v] : acc0[j][v];
        if (BIAS) x += bias[col];
        if (ACT_SILU) x = silu(x);
        int64_t o = (int64_t)(rbase + v) * ldc + col;
        if (RES) x += Resid[o];
        C[o] = x;
      }
    }
  }
}

// =====================================================================
// Retention on pre-packed bf16 operands:
// Qpk/Kpk: [bh][S][HD/2] dwords.  VT: [bh][HD][S/2] dwords (t-pairs).
// O = (Q K^T .* D) V, D[s,t]=gamma_h^(s-t) for s>=t. 64-row s-block per
// workgroup, async-staged causal t-tiles. grid=(S/64, B*H), 256 thr.
// =====================================================================
__global__ __launch_bounds__(256) void retention_kernel(
    const unsigned int* __restrict__ Qpk, const unsigned int* __restrict__ Kpk,
    const unsigned int* __restrict__ VT, float* __restrict__ O, int S) {
  __shared__ unsigned int lsQ[64 * 64];       // [s][hd-pair]   16 KB
  __shared__ unsigned int lsKT[64 * 64];      // [t][hd-pair]   16 KB
  __shared__ unsigned int lsVT[HEADDIM * 32]; // [hd][t-pair]   16 KB
  __shared__ unsigned int lsS[64 * 32];       // [s][t-pair]     8 KB

  const int bh = blockIdx.y;
  const int h = bh & (NHEADS - 1);
  const unsigned int* Qg = Qpk + (int64_t)bh * S * (HEADDIM / 2);
  const unsigned int* Kg = Kpk + (int64_t)bh * S * (HEADDIM / 2);
  const unsigned int* Vg = VT + (int64_t)bh * HEADDIM * (S / 2);
  float* Og = O + (int64_t)bh * S * HEADDIM;

  const int s0 = blockIdx.x * 64;
  const int tid = threadIdx.x, lane = tid & 31, wave = tid >> 5;

  const unsigned qoff = lds_off_of(lsQ);
  const unsigned koff = lds_off_of(lsKT);
  const unsigned voff = lds_off_of(lsVT);

  // gamma_h = 1 - exp(linspace(log(1/32), log(1/512), H))
  const float lg0 = logf(1.f / 32.f), lg1 = logf(1.f / 512.f);
  const float gamma = 1.f - expf(lg0 + (lg1 - lg0) * ((float)h / (NHEADS - 1)));
  const float l2g = log2f(gamma);

  // pin Q tile in LDS: 64 rows x 64 dwords, async
#pragma unroll
  for (int pass = 0; pass < 4; ++pass) {
    int idx = tid + pass * 256;
    int r = idx >> 4, ch = (idx & 15) << 2;
    async_ld128(qoff + (unsigned)((r * 64 + ch) << 2),
                Qg + (int64_t)(s0 + r) * 64 + ch);
  }

  const int wm = (wave & 3) * 16;  // row tile
  const int wh = (wave >> 2);      // 0/1 : which half of t / hd
  v8f oacc[4] = {};                // hd tiles at wh*64 + j*16
  async_wait0();
  __syncthreads();

  for (int t0 = 0; t0 <= s0; t0 += 64) {
    // K tile: [t][hd-pair] (natural packed layout), 64x64 dwords
#pragma unroll
    for (int pass = 0; pass < 4; ++pass) {
      int idx = tid + pass * 256;
      int r = idx >> 4, ch = (idx & 15) << 2;
      async_ld128(koff + (unsigned)((r * 64 + ch) << 2),
                  Kg + (int64_t)(t0 + r) * 64 + ch);
    }
    // V^T tile: [hd][t-pair], 128x32 dwords
#pragma unroll
    for (int pass = 0; pass < 4; ++pass) {
      int idx = tid + pass * 256;
      int r = idx >> 3, ch = (idx & 7) << 2;
      async_ld128(voff + (unsigned)((r * 32 + ch) << 2),
                  Vg + (int64_t)r * (S / 2) + (t0 >> 1) + ch);
    }
    async_wait0();
    __syncthreads();

    // ---- stage 1: S = Q * K^T (64x64), K-dim = HEADDIM ----
    v8f sa = {}, sb = {};
#pragma unroll
    for (int kk = 0; kk < HEADDIM; kk += 32) {
      v16bf aq = frag_A(lsQ + wm * 64, 64, kk >> 1, lane);
      v16bf k0f = frag_B(lsKT + (wh * 32) * 64, 64, kk >> 1, lane);
      v16bf k1f = frag_B(lsKT + (wh * 32 + 16) * 64, 64, kk >> 1, lane);
      sa = wmma_bf16(aq, k0f, sa);
      sb = wmma_bf16(aq, k1f, sb);
    }
    // decay mask + restage as bf16 A-operand for stage 2
    {
      unsigned short* sh = (unsigned short*)lsS;
      const int srow = wm + ((lane >> 4) << 3);
      const int tc = wh * 32 + (lane & 15);
#pragma unroll
      for (int v = 0; v < 8; ++v) {
        int s = s0 + srow + v;
        int ta = t0 + tc, tb = t0 + tc + 16;
        float ma = (s >= ta) ? exp2f((float)(s - ta) * l2g) : 0.f;
        float mb = (s >= tb) ? exp2f((float)(s - tb) * l2g) : 0.f;
        sh[(srow + v) * 64 + tc] = bf16bits(sa[v] * ma);
        sh[(srow + v) * 64 + tc + 16] = bf16bits(sb[v] * mb);
      }
    }
    __syncthreads();

    // ---- stage 2: O += S * V (64x64 x 64x128) ----
#pragma unroll
    for (int kk = 0; kk < 64; kk += 32) {
      v16bf as = frag_A(lsS + wm * 32, 32, kk >> 1, lane);
#pragma unroll
      for (int j = 0; j < 4; ++j) {
        v16bf bv = frag_B(lsVT + (wh * 64 + j * 16) * 32, 32, kk >> 1, lane);
        oacc[j] = wmma_bf16(as, bv, oacc[j]);
      }
    }
    __syncthreads();
  }

  // write O (B,H,S,HD) in f32
  const int colb = wh * 64 + (lane & 15);
  const int rb = s0 + wm + ((lane >> 4) << 3);
#pragma unroll
  for (int j = 0; j < 4; ++j)
#pragma unroll
    for (int v = 0; v < 8; ++v)
      Og[(int64_t)(rb + v) * HEADDIM + colb + j * 16] = oacc[j][v];
}

// =====================================================================
// LayerNorm over DMODEL, emitting packed bf16 pairs (GEMM A-operand form).
// One block (256 thr) per row; each thread owns pairs tid and tid+256.
// =====================================================================
__global__ __launch_bounds__(256) void layernorm_pack_kernel(
    const float* __restrict__ X, const float* __restrict__ w,
    const float* __restrict__ b, unsigned int* __restrict__ Yp) {
  __shared__ float r1[256];
  __shared__ float r2[256];
  const int64_t row = blockIdx.x;
  const float* x = X + row * DMODEL;
  unsigned int* y = Yp + row * (DMODEL / 2);
  const int tid = threadIdx.x;
  float v[4], s = 0.f, ss = 0.f;
#pragma unroll
  for (int e = 0; e < 4; ++e) {
    int c = ((e >> 1) * 256 + tid) * 2 + (e & 1);
    v[e] = x[c];
    s += v[e];
    ss += v[e] * v[e];
  }
  r1[tid] = s;
  r2[tid] = ss;
  __syncthreads();
  for (int o = 128; o > 0; o >>= 1) {
    if (tid < o) { r1[tid] += r1[tid + o]; r2[tid] += r2[tid + o]; }
    __syncthreads();
  }
  float mu = r1[0] * (1.f / DMODEL);
  float var = r2[0] * (1.f / DMODEL) - mu * mu;
  float rstd = rsqrtf(var + 1e-5f);
#pragma unroll
  for (int pr = 0; pr < 2; ++pr) {
    int p = pr * 256 + tid;
    float lo = (v[pr * 2] - mu) * rstd * w[2 * p] + b[2 * p];
    float hi = (v[pr * 2 + 1] - mu) * rstd * w[2 * p + 1] + b[2 * p + 1];
    y[p] = pack_bf16(lo, hi);
  }
}

// =====================================================================
// GroupNorm over HEADDIM per head, fused with (B,H,S,HD)->(B,S,D) transpose.
// =====================================================================
__global__ __launch_bounds__(256) void groupnorm_kernel(
    const float* __restrict__ YH, const float* __restrict__ w,
    const float* __restrict__ b, float* __restrict__ YG, int S) {
  const int bs = blockIdx.x;
  const int batch = bs / S, s = bs % S;
  const int tid = threadIdx.x, lane = tid & 31, h = tid >> 5;
  const float* src = YH + ((int64_t)(batch * NHEADS + h) * S + s) * HEADDIM;
  float v[4], sum = 0.f, ss = 0.f;
#pragma unroll
  for (int e = 0; e < 4; ++e) {
    v[e] = src[lane + e * 32];
    sum += v[e];
    ss += v[e] * v[e];
  }
#pragma unroll
  for (int o = 16; o > 0; o >>= 1) {
    sum += __shfl_xor(sum, o, 32);
    ss += __shfl_xor(ss, o, 32);
  }
  float mu = sum * (1.f / HEADDIM);
  float var = ss * (1.f / HEADDIM) - mu * mu;
  float rstd = rsqrtf(var + 1e-5f);
  float* dst = YG + ((int64_t)batch * S + s) * DMODEL + h * HEADDIM;
  const float* wg = w + h * HEADDIM;
  const float* bg = b + h * HEADDIM;
#pragma unroll
  for (int e = 0; e < 4; ++e) {
    int c = lane + e * 32;
    dst[c] = (v[e] - mu) * rstd * wg[c] + bg[c];
  }
}

// =====================================================================
// xPos rotary: f32 (B,H,S,HD) in -> packed bf16 pairs out.
// expo_sign = +1 for Q (x scale), -1 for K (/ scale).
// =====================================================================
__global__ __launch_bounds__(256) void xpos_rotary_pack_kernel(
    const float* __restrict__ P, unsigned int* __restrict__ dst, int S,
    float expo_sign) {
  int64_t idx = (int64_t)blockIdx.x * 256 + threadIdx.x;
  int j = (int)(idx & 63);  // pair index within head dim
  int64_t rowi = idx >> 6;  // (b*H + h)*S + s
  int s = (int)(rowi % S);
  const float* p = P + rowi * HEADDIM + (j << 1);
  float sbase = ((float)(2 * j) + 0.4f * HEADDIM) / (1.4f * HEADDIM);
  float power = (float)s * (1.f / 512.f);
  float sc = __powf(sbase, expo_sign * power);
  float inv_freq = __powf(10000.f, -(float)j / (float)(HEADDIM / 2));
  float ang = (float)s * inv_freq;
  float sn, cs;
  __sincosf(ang, &sn, &cs);
  sn *= sc;
  cs *= sc;
  float x0 = p[0], x1 = p[1];
  dst[rowi * (HEADDIM / 2) + j] = pack_bf16(x0 * cs - x1 * sn,
                                            x1 * cs + x0 * sn);
}

// elementwise product, packed bf16 output (one dword pair per thread)
__global__ __launch_bounds__(256) void mul_pack_kernel(
    const float* __restrict__ a, const float* __restrict__ b,
    unsigned int* __restrict__ outp, int64_t ndw) {
  int64_t i = (int64_t)blockIdx.x * 256 + threadIdx.x;
  if (i < ndw)
    outp[i] = pack_bf16(a[2 * i] * b[2 * i], a[2 * i + 1] * b[2 * i + 1]);
}

// flat f32 -> packed bf16 pairs (contiguous rows)
__global__ __launch_bounds__(256) void cast_pack_kernel(
    const float* __restrict__ in, unsigned int* __restrict__ outp,
    int64_t ndw) {
  int64_t i = (int64_t)blockIdx.x * 256 + threadIdx.x;
  if (i < ndw) outp[i] = pack_bf16(in[2 * i], in[2 * i + 1]);
}

// transpose + pack: in f32 [K][N] (ld = N) -> out [N][K/2] dwords.
// grid.y = batch (in += y*strideIn, out += y*strideOut)
__global__ __launch_bounds__(256) void castT_pack_kernel(
    const float* __restrict__ in, int64_t strideIn,
    unsigned int* __restrict__ outp, int64_t strideOut, int K, int N) {
  const int khalf = K >> 1;
  const int64_t per = (int64_t)N * khalf;
  int64_t i = (int64_t)blockIdx.x * 256 + threadIdx.x;
  if (i >= per) return;
  int n = (int)(i / khalf);
  int kp = (int)(i % khalf);
  const float* src = in + (int64_t)blockIdx.y * strideIn;
  unsigned int* dst = outp + (int64_t)blockIdx.y * strideOut;
  dst[i] = pack_bf16(src[(int64_t)(kp << 1) * N + n],
                     src[((int64_t)(kp << 1) + 1) * N + n]);
}

// =====================================================================
// Launcher. Workspace: 44M 4-byte units = 176 MB.
// =====================================================================
extern "C" void kernel_launch(void* const* d_in, const int* in_sizes, int n_in,
                              void* d_out, int out_size, void* d_ws,
                              size_t ws_size, hipStream_t stream) {
  (void)in_sizes; (void)n_in; (void)out_size; (void)ws_size;
  const float* X = (const float*)d_in[0];
  const float* ln1_w = (const float*)d_in[1];
  const float* ln1_b = (const float*)d_in[2];
  const float* ln2_w = (const float*)d_in[3];
  const float* ln2_b = (const float*)d_in[4];
  const float* WQ = (const float*)d_in[5];
  const float* WK = (const float*)d_in[6];
  const float* WV = (const float*)d_in[7];
  const float* gn_w = (const float*)d_in[8];
  const float* gn_b = (const float*)d_in[9];
  const float* WG = (const float*)d_in[10];
  const float* WO = (const float*)d_in[11];
  const float* fw1 = (const float*)d_in[12];
  const float* fb1 = (const float*)d_in[13];
  const float* fw2 = (const float*)d_in[14];
  const float* fb2 = (const float*)d_in[15];
  float* out = (float*)d_out;

  const size_t MU = (size_t)1024 * 1024;  // 1M 4-byte units
  char* wsb = (char*)d_ws;
  auto F = [&](size_t off) { return (float*)(wsb + off * 4); };
  auto U = [&](size_t off) { return (unsigned int*)(wsb + off * 4); };

  float* Qb = F(0 * MU);          // 4M f32: Q
  float* Kb = F(4 * MU);          // 4M f32: K; reused as groupnorm out
  float* Vb = F(8 * MU);          // 4M f32: V; reused as Y residual stream
  float* YH = F(12 * MU);         // 4M f32: attention out (B,H,S,HD)
  float* Gb = F(16 * MU);         // 4M f32: gate
  float* X1 = F(20 * MU);         // 4M f32: inter-layer activations
  float* H1 = F(24 * MU);         // 4M f32: FFN hidden chunk (1024x4096)
  unsigned int* XNh = U(28 * MU); // 2M dw: LN1 out packed [4096][512]
  unsigned int* Ah = U(30 * MU);  // 2M dw: GY packed, then YN packed
  unsigned int* Hh = U(32 * MU);  // 2M dw: H1 chunk packed [1024][2048]
  unsigned int* Qpk = U(34 * MU); // 2M dw: rotated Q packed
  unsigned int* Kpk = U(36 * MU); // 2M dw: rotated K packed
  unsigned int* VTpk = U(38 * MU);// 2M dw: V^T packed [bh][128][1024]
  unsigned int* Wh1 = U(40 * MU); // 2M dw: weight scratch (B^T packed)
  unsigned int* Wh2 = U(42 * MU); // 2M dw: weight scratch (B^T packed)

  const int64_t ZERO = 0;

  for (int i = 0; i < L_LAYERS; ++i) {
    const float* Xin = (i == 0) ? X : X1;
    float* Xnext = (i == 0) ? X1 : out;

    // LN1 -> packed A operand
    layernorm_pack_kernel<<<MROWS, 256, 0, stream>>>(
        Xin, ln1_w + i * DMODEL, ln1_b + i * DMODEL, XNh);

    // Q/K/V projections: per (b,h) GEMM [2048 x 128 x 1024]
    {
      dim3 g(HEADDIM / 128, SEQ / 128, BATCH * NHEADS);
      const int64_t sA = (int64_t)SEQ * (DMODEL / 2);
      const int64_t sB = (int64_t)HEADDIM * (DMODEL / 2);
      const int64_t sC = (int64_t)SEQ * HEADDIM;
      const size_t lw = (size_t)i * NHEADS * DMODEL * HEADDIM;
      const float* Wsrc[3] = {WQ + lw, WK + lw, WV + lw};
      float* Cdst[3] = {Qb, Kb, Vb};
      for (int q = 0; q < 3; ++q) {
        castT_pack_kernel<<<dim3(256, NHEADS), 256, 0, stream>>>(
            Wsrc[q], (int64_t)DMODEL * HEADDIM, Wh1, sB, DMODEL, HEADDIM);
        gemm_pk<0, 0, 0><<<g, 256, 0, stream>>>(
            XNh, sA, NHEADS, DMODEL / 2, Wh1, sB, NHEADS, DMODEL / 2, nullptr,
            nullptr, ZERO, Cdst[q], sC, SEQ, HEADDIM, DMODEL, HEADDIM);
      }
    }

    // xPos rotary -> packed Q/K
    {
      const int64_t pairs = (int64_t)BATCH * NHEADS * SEQ * (HEADDIM / 2);
      const int blocks = (int)(pairs / 256);
      xpos_rotary_pack_kernel<<<blocks, 256, 0, stream>>>(Qb, Qpk, SEQ, +1.f);
      xpos_rotary_pack_kernel<<<blocks, 256, 0, stream>>>(Kb, Kpk, SEQ, -1.f);
    }
    // V -> V^T packed: per bh, [S][HD] -> [HD][S/2]
    castT_pack_kernel<<<dim3(512, BATCH * NHEADS), 256, 0, stream>>>(
        Vb, (int64_t)SEQ * HEADDIM, VTpk, (int64_t)HEADDIM * (SEQ / 2), SEQ,
        HEADDIM);

    // retention -> YH (f32, (B,H,S,HD))
    retention_kernel<<<dim3(SEQ / 64, BATCH * NHEADS), 256, 0, stream>>>(
        Qpk, Kpk, VTpk, YH, SEQ);

    // groupnorm + transpose to (B,S,D) -> Kb (f32)
    groupnorm_kernel<<<MROWS, 256, 0, stream>>>(YH, gn_w + i * DMODEL,
                                                gn_b + i * DMODEL, Kb, SEQ);

    // G = silu(XN @ WG)  [4096 x 1024 x 1024] -> Gb (f32)
    castT_pack_kernel<<<dim3(2048, 1), 256, 0, stream>>>(
        WG + (size_t)i * DMODEL * DMODEL, ZERO, Wh1,
        (int64_t)DMODEL * (DMODEL / 2), DMODEL, DMODEL);
    gemm_pk<0, 1, 0><<<dim3(DMODEL / 128, MROWS / 128, 1), 256, 0, stream>>>(
        XNh, ZERO, 1, DMODEL / 2, Wh1, ZERO, 1, DMODEL / 2, nullptr, nullptr,
        ZERO, Gb, ZERO, MROWS, DMODEL, DMODEL, DMODEL);

    // GY = G * Ygn -> packed Ah
    mul_pack_kernel<<<8192, 256, 0, stream>>>(Gb, Kb, Ah,
                                              (int64_t)MROWS * (DMODEL / 2));

    // Y = GY @ WO + Xin -> Vb (f32)
    castT_pack_kernel<<<dim3(2048, 1), 256, 0, stream>>>(
        WO + (size_t)i * DMODEL * DMODEL, ZERO, Wh1,
        (int64_t)DMODEL * (DMODEL / 2), DMODEL, DMODEL);
    gemm_pk<0, 0, 1><<<dim3(DMODEL / 128, MROWS / 128, 1), 256, 0, stream>>>(
        Ah, ZERO, 1, DMODEL / 2, Wh1, ZERO, 1, DMODEL / 2, nullptr, Xin, ZERO,
        Vb, ZERO, MROWS, DMODEL, DMODEL, DMODEL);

    // LN2: Vb -> packed Ah (Yn)
    layernorm_pack_kernel<<<MROWS, 256, 0, stream>>>(
        Vb, ln2_w + i * DMODEL, ln2_b + i * DMODEL, Ah);

    // FFN weights -> packed transposed
    castT_pack_kernel<<<dim3(8192, 1), 256, 0, stream>>>(
        fw1 + (size_t)i * DMODEL * FFNDIM, ZERO, Wh1,
        (int64_t)FFNDIM * (DMODEL / 2), DMODEL, FFNDIM);
    castT_pack_kernel<<<dim3(8192, 1), 256, 0, stream>>>(
        fw2 + (size_t)i * FFNDIM * DMODEL, ZERO, Wh2,
        (int64_t)DMODEL * (FFNDIM / 2), FFNDIM, DMODEL);

    // FFN, chunked over rows (1024 at a time)
    const int CH = 1024;
    for (int c0 = 0; c0 < MROWS; c0 += CH) {
      gemm_pk<1, 1, 0><<<dim3(FFNDIM / 128, CH / 128, 1), 256, 0, stream>>>(
          Ah + (size_t)c0 * (DMODEL / 2), ZERO, 1, DMODEL / 2, Wh1, ZERO, 1,
          DMODEL / 2, fb1 + (size_t)i * FFNDIM, nullptr, ZERO, H1, ZERO, CH,
          FFNDIM, DMODEL, FFNDIM);
      cast_pack_kernel<<<8192, 256, 0, stream>>>(
          H1, Hh, (int64_t)CH * (FFNDIM / 2));
      gemm_pk<1, 0, 1><<<dim3(DMODEL / 128, CH / 128, 1), 256, 0, stream>>>(
          Hh, ZERO, 1, FFNDIM / 2, Wh2, ZERO, 1, FFNDIM / 2,
          fb2 + (size_t)i * DMODEL, Vb + (size_t)c0 * DMODEL, ZERO,
          Xnext + (size_t)c0 * DMODEL, ZERO, CH, DMODEL, FFNDIM, DMODEL);
    }
  }
}
